// GraphSAGEEncoder_377957122578
// MI455X (gfx1250) — compile-verified, matrix operand order checked by reading, not run on hardware
//
#include <hip/hip_runtime.h>
#include <hip/hip_bf16.h>

typedef __attribute__((ext_vector_type(16))) __bf16 v16bf;
typedef __attribute__((ext_vector_type(8)))  float  v8f;

#define FEAT 64

// ---------------------------------------------------------------------------
// Degree accumulation: one thread per edge.
// ---------------------------------------------------------------------------
__global__ void sage_degree_kernel(const int* __restrict__ dst,
                                   float* __restrict__ deg, int nedges) {
    int e = blockIdx.x * blockDim.x + threadIdx.x;
    if (e < nedges) atomicAdd(&deg[dst[e]], 1.0f);
}

// ---------------------------------------------------------------------------
// Edge scatter: agg[dst] += feat[src].  64 lanes per edge (one feature each),
// 4 edges per 256-thread block. Coalesced 256B row reads, f32 atomics out.
// ---------------------------------------------------------------------------
__global__ void sage_scatter_kernel(const int* __restrict__ src,
                                    const int* __restrict__ dst,
                                    const float* __restrict__ feat,
                                    float* __restrict__ agg, int nedges) {
    int e = blockIdx.x * 4 + (threadIdx.x >> 6);
    int f = threadIdx.x & 63;
    if (e < nedges) {
        int s = src[e];
        int d = dst[e];
        atomicAdd(&agg[(size_t)d * FEAT + f], feat[(size_t)s * FEAT + f]);
    }
}

// ---------------------------------------------------------------------------
// Fused SAGE layer:  out = relu?( [agg/deg | x] @ [Wl;Wr] + b )
// One wave per 16x64 output tile; 8 waves (128 rows) per block.
// Combined K = 128 -> 4 WMMA K-steps of 32; 4 column tiles of 16.
// Weights converted to bf16 once per block into LDS, pre-swizzled into the
// exact B-fragment layout so each fragment is one contiguous 32B LDS read.
// ---------------------------------------------------------------------------
__global__ void __launch_bounds__(256)
sage_fused_wmma_kernel(const float* __restrict__ agg,
                       const float* __restrict__ deg,
                       const float* __restrict__ xin,
                       const float* __restrict__ Wl,   // 64x64 (agg side)
                       const float* __restrict__ Wr,   // 64x64 (x side)
                       const float* __restrict__ bias, // 64
                       float* __restrict__ out,
                       int nrows, int relu_flag) {
    // Swizzled bf16 weights: [ks][ct][lane][16 elems] -> 8192 bf16 = 16 KB
    __shared__ __align__(32) __bf16 sW[4 * 4 * 32 * 16];
    __shared__ float sBias[FEAT];

    const int tid = threadIdx.x;

    // Fill LDS: for each (ks, ct, lane, e) compute (k, n) per the 16-bit
    // B-matrix lane layout and stash combined weight W[k][n] as bf16.
    for (int i = tid; i < 8192; i += 256) {
        int e    = i & 15;
        int lane = (i >> 4) & 31;
        int ct   = (i >> 9) & 3;
        int ks   = (i >> 11) & 3;
        int v = e >> 1, j = e & 1;
        int k = ks * 32 + ((v & 4) ? 16 : 0) + (lane >> 4) * 8 + 2 * (v & 3) + j;
        int n = ct * 16 + (lane & 15);
        float w = (k < 64) ? Wl[k * FEAT + n] : Wr[(k - 64) * FEAT + n];
        sW[i] = (__bf16)w;
    }
    if (tid < FEAT) sBias[tid] = bias[tid];
    __syncthreads();

    const int wave  = tid >> 5;
    const int lane  = tid & 31;
    const int tile  = blockIdx.x * 8 + wave;      // wave-uniform
    const int ntile = (nrows + 15) >> 4;
    if (tile >= ntile) return;                    // wave-granular: EXEC stays all-1

    const int rbase = tile * 16;
    const int mrow  = rbase + (lane & 15);
    const int khalf = lane >> 4;

    const float invdeg = 1.0f / fmaxf(deg[mrow], 1.0f);

    v8f acc[4] = {};

#pragma unroll
    for (int ks = 0; ks < 4; ++ks) {
        // ---- A fragment (16x32 bf16): lane holds row mrow, two 8-wide K
        // chunks (lo: kb, hi: kb+16 region, split by khalf). ks<2 reads the
        // normalized aggregate, ks>=2 reads x (the concat's second half).
        v16bf afrag;
#pragma unroll
        for (int c = 0; c < 2; ++c) {
            int kb = ks * 32 + c * 16 + khalf * 8;   // 8 contiguous floats
            float4 p0, p1;
            if (ks < 2) {
                const float* rp = agg + (size_t)mrow * FEAT + kb;
                p0 = *(const float4*)(rp);
                p1 = *(const float4*)(rp + 4);
                p0.x *= invdeg; p0.y *= invdeg; p0.z *= invdeg; p0.w *= invdeg;
                p1.x *= invdeg; p1.y *= invdeg; p1.z *= invdeg; p1.w *= invdeg;
            } else {
                const float* rp = xin + (size_t)mrow * FEAT + (kb - 64);
                p0 = *(const float4*)(rp);
                p1 = *(const float4*)(rp + 4);
            }
            int eb = 8 * c;
            afrag[eb + 0] = (__bf16)p0.x; afrag[eb + 1] = (__bf16)p0.y;
            afrag[eb + 2] = (__bf16)p0.z; afrag[eb + 3] = (__bf16)p0.w;
            afrag[eb + 4] = (__bf16)p1.x; afrag[eb + 5] = (__bf16)p1.y;
            afrag[eb + 6] = (__bf16)p1.z; afrag[eb + 7] = (__bf16)p1.w;
        }

#pragma unroll
        for (int ct = 0; ct < 4; ++ct) {
            const v16bf bfrag =
                *(const v16bf*)(sW + (((ks * 4 + ct) * 32 + lane) << 4));
            acc[ct] = __builtin_amdgcn_wmma_f32_16x16x32_bf16(
                false, afrag, false, bfrag, (short)0, acc[ct], false, false);
        }
    }

    // ---- Epilogue: C/D layout -> VGPR r holds row rbase + r + 8*khalf,
    // column ct*16 + (lane&15). Add bias, optional ReLU, store f32.
#pragma unroll
    for (int ct = 0; ct < 4; ++ct) {
        int col = ct * 16 + (lane & 15);
        float b = sBias[col];
#pragma unroll
        for (int r = 0; r < 8; ++r) {
            int row = rbase + r + 8 * khalf;
            float val = acc[ct][r] + b;
            if (relu_flag) val = fmaxf(val, 0.0f);
            out[(size_t)row * FEAT + col] = val;
        }
    }
}

// ---------------------------------------------------------------------------
// Host launcher.
// Inputs: x, edge_index, W1l, W1r, b1, W2l, W2r, b2
// Workspace layout (floats): [deg (padded) | agg1 (N*64) | agg2 (N*64)]
// h (layer-1 activations) lives in d_out and is overwritten in place by the
// final GEMM (each wave only reads its own rows of h before storing them).
// ---------------------------------------------------------------------------
extern "C" void kernel_launch(void* const* d_in, const int* in_sizes, int n_in,
                              void* d_out, int out_size, void* d_ws, size_t ws_size,
                              hipStream_t stream) {
    const float* x   = (const float*)d_in[0];
    const int*   ei  = (const int*)d_in[1];
    const float* W1l = (const float*)d_in[2];
    const float* W1r = (const float*)d_in[3];
    const float* b1  = (const float*)d_in[4];
    const float* W2l = (const float*)d_in[5];
    const float* W2r = (const float*)d_in[6];
    const float* b2  = (const float*)d_in[7];
    float* out = (float*)d_out;

    const int n = in_sizes[0] / FEAT;   // 50000 nodes
    const int E = in_sizes[1] / 2;      // 800000 edges
    const int* src = ei;
    const int* dst = ei + E;

    float* ws = (float*)d_ws;
    size_t degPad = ((size_t)n + 255) & ~(size_t)255;
    float* deg  = ws;
    float* agg1 = ws + degPad;
    float* agg2 = agg1 + (size_t)n * FEAT;
    float* h    = out;  // layer-1 activations live in d_out

    // Zero deg + agg1 + agg2 in one stream-ordered memset (capture-legal).
    hipMemsetAsync(ws, 0, (degPad + 2 * (size_t)n * FEAT) * sizeof(float), stream);

    const int tiles  = (n + 15) / 16;   // 3125 exactly
    const int blocks = (tiles + 7) / 8;

    sage_degree_kernel<<<(E + 255) / 256, 256, 0, stream>>>(dst, deg, E);
    sage_scatter_kernel<<<(E + 3) / 4, 256, 0, stream>>>(src, dst, x, agg1, E);
    sage_fused_wmma_kernel<<<blocks, 256, 0, stream>>>(
        agg1, deg, x, W1l, W1r, b1, h, n, /*relu=*/1);
    sage_scatter_kernel<<<(E + 3) / 4, 256, 0, stream>>>(src, dst, h, agg2, E);
    sage_fused_wmma_kernel<<<blocks, 256, 0, stream>>>(
        agg2, deg, h, W2l, W2r, b2, out, n, /*relu=*/0);
}
